// LinearCrossAttention_12601434046877
// MI455X (gfx1250) — matrix operand order repaired and checked by reading, not compiled, not run on hardware
//
#include <hip/hip_runtime.h>
#include <hip/hip_bf16.h>

typedef __attribute__((ext_vector_type(16))) __bf16 v16bf;
typedef __attribute__((ext_vector_type(8)))  __bf16 v8bf;
typedef __attribute__((ext_vector_type(8)))  float  v8f;

// ---------------------------------------------------------------------------
// Transpose fp32 [R x C] -> bf16 [C x R]   (coalesced reads)
// ---------------------------------------------------------------------------
__global__ __launch_bounds__(256) void transpose_f32_to_bf16_kernel(
    const float* __restrict__ in, __bf16* __restrict__ out, int R, int C) {
  long long i = (long long)blockIdx.x * 256 + threadIdx.x;
  if (i < (long long)R * C) {
    int r = (int)(i / C), c = (int)(i % C);
    out[(long long)c * R + r] = (__bf16)in[i];
  }
}

// ---------------------------------------------------------------------------
// v: fp32 [B][N=512][D=1024] -> bf16 vt [B*H][128(l)][512(n)]
// ---------------------------------------------------------------------------
__global__ __launch_bounds__(256) void v_transpose_kernel(
    const float* __restrict__ in, __bf16* __restrict__ out) {
  long long i = (long long)blockIdx.x * 256 + threadIdx.x;  // 2097152 total
  int b   = (int)(i >> 19);
  int rem = (int)(i & 524287);
  int n   = rem >> 10;
  int h   = (rem >> 7) & 7;
  int l   = rem & 127;
  out[(((long long)(b * 8 + h) * 128 + l) << 9) + n] = (__bf16)in[i];
}

// ---------------------------------------------------------------------------
// attn: fp32 [Z=32][128(d)][128(l)] -> bf16 [Z][128(l)][128(d)]
// ---------------------------------------------------------------------------
__global__ __launch_bounds__(256) void attn_transpose_kernel(
    const float* __restrict__ in, __bf16* __restrict__ out) {
  int i = blockIdx.x * 256 + threadIdx.x;   // 524288 total
  int z = i >> 14, rem = i & 16383;
  int d = rem >> 7, l = rem & 127;
  out[(z << 14) + (l << 7) + d] = (__bf16)in[i];
}

// ---------------------------------------------------------------------------
// LayerNorm over last dim -> bf16.  One block per row, 256 threads, float4 IO.
// ---------------------------------------------------------------------------
__global__ __launch_bounds__(256) void layernorm_bf16_kernel(
    const float* __restrict__ x, const float* __restrict__ gamma,
    const float* __restrict__ beta, __bf16* __restrict__ out, int len) {
  const long long row = blockIdx.x;
  const float* xr = x + row * (long long)len;
  __bf16* orow = out + row * (long long)len;
  int tid = threadIdx.x, lane = tid & 31, wave = tid >> 5;

  float s = 0.f, s2 = 0.f;
  for (int i = tid * 4; i < len; i += 256 * 4) {
    float4 v = *(const float4*)(xr + i);
    s  += v.x + v.y + v.z + v.w;
    s2 += v.x * v.x + v.y * v.y + v.z * v.z + v.w * v.w;
  }
#pragma unroll
  for (int off = 16; off > 0; off >>= 1) {
    s  += __shfl_xor(s, off);
    s2 += __shfl_xor(s2, off);
  }
  __shared__ float red[2][8];
  if (lane == 0) { red[0][wave] = s; red[1][wave] = s2; }
  __syncthreads();
  float ts = 0.f, ts2 = 0.f;
#pragma unroll
  for (int w = 0; w < 8; ++w) { ts += red[0][w]; ts2 += red[1][w]; }
  float mu  = ts / (float)len;
  float var = ts2 / (float)len - mu * mu;
  float inv = rsqrtf(var + 1e-5f);

  for (int i = tid * 4; i < len; i += 256 * 4) {
    float4 v = *(const float4*)(xr + i);
    float4 g = *(const float4*)(gamma + i);
    float4 b = *(const float4*)(beta + i);
    orow[i + 0] = (__bf16)((v.x - mu) * inv * g.x + b.x);
    orow[i + 1] = (__bf16)((v.y - mu) * inv * g.y + b.y);
    orow[i + 2] = (__bf16)((v.z - mu) * inv * g.z + b.z);
    orow[i + 3] = (__bf16)((v.w - mu) * inv * g.w + b.w);
  }
}

// ---------------------------------------------------------------------------
// Softmax over contiguous 128-wide segments (head dim). One wave per segment.
// ---------------------------------------------------------------------------
__global__ __launch_bounds__(256) void softmax_hd_kernel(
    const float* __restrict__ in, __bf16* __restrict__ out) {
  long long seg = (long long)blockIdx.x * 8 + (threadIdx.x >> 5);
  int lane = threadIdx.x & 31;
  const float* p = in + seg * 128 + lane * 4;
  float4 v = *(const float4*)p;
  float m = fmaxf(fmaxf(v.x, v.y), fmaxf(v.z, v.w));
#pragma unroll
  for (int off = 16; off > 0; off >>= 1) m = fmaxf(m, __shfl_xor(m, off));
  float e0 = __expf(v.x - m), e1 = __expf(v.y - m);
  float e2 = __expf(v.z - m), e3 = __expf(v.w - m);
  float s = e0 + e1 + e2 + e3;
#pragma unroll
  for (int off = 16; off > 0; off >>= 1) s += __shfl_xor(s, off);
  float r = 1.f / s;
  __bf16* o = out + seg * 128 + lane * 4;
  o[0] = (__bf16)(e0 * r); o[1] = (__bf16)(e1 * r);
  o[2] = (__bf16)(e2 * r); o[3] = (__bf16)(e3 * r);
}

// ---------------------------------------------------------------------------
// Column softmax of k over N=512 rows, fused transpose:
//   in : fp32 [B][512][1024]   (col = h*128+d)
//   out: bf16 kt [B*H][128(d)][512(n)]
// One thread per column; adjacent threads -> adjacent columns (coalesced reads).
// grid = (1024/256, B)
// ---------------------------------------------------------------------------
__global__ __launch_bounds__(256) void softmax_col_kernel(
    const float* __restrict__ in, __bf16* __restrict__ out) {
  int b = blockIdx.y;
  int col = blockIdx.x * 256 + threadIdx.x;            // 0..1023
  const float* p = in + (long long)b * 512 * 1024 + col;
  int h = col >> 7, d = col & 127;
  __bf16* o = out + (((long long)(b * 8 + h) * 128 + d) << 9);  // *512
  float m = -3.0e38f;
  for (int r = 0; r < 512; ++r) m = fmaxf(m, p[(long long)r * 1024]);
  float s = 0.f;
  for (int r = 0; r < 512; ++r) s += __expf(p[(long long)r * 1024] - m);
  float inv = 1.f / s;
  for (int r = 0; r < 512; ++r)
    o[r] = (__bf16)(__expf(p[(long long)r * 1024] - m) * inv);
}

// ---------------------------------------------------------------------------
// Batched bf16 WMMA GEMM (NT form): C[z] = A[z] * Bt[z]^T (+bias), fp32 out.
//   A : [M x K] row-major (lda)        Bt : [N x K] row-major (ldbt)
// Batch offsets (z = blockIdx.z): off = (z / zDiv) * Z1 + (z % zDiv) * Z2.
// Block: 256 thr = 8 waves; block tile 128x128; wave tile 32x64 (2x4 of 16x16).
// K-chunk = 32 staged into LDS with GLOBAL_LOAD_ASYNC_TO_LDS_B128 (ASYNCcnt),
// fragments via ds_load_b128 per the v_wmma_f32_16x16x32_bf16 lane maps:
//   A frag: lanes 0-15 K{0-7,16-23}, lanes 16-31 K{8-15,24-31}
//   B frag: lanes 0-15 K{0-7,16-23}, lanes 16-31 K{8-15,24-31}
// ---------------------------------------------------------------------------
__global__ __launch_bounds__(256) void gemm_bf16_nt_kernel(
    const __bf16* __restrict__ A, const __bf16* __restrict__ Bt,
    float* __restrict__ C, const float* __restrict__ bias,
    int K, int lda, int ldbt, int ldc, int zDiv,
    long long aZ1, long long aZ2, long long bZ1, long long bZ2,
    long long cZ1, long long cZ2) {
  __shared__ __bf16 lA[128][48];   // [m][k], pitch 48 (16B-aligned rows)
  __shared__ __bf16 lB[128][48];   // [n][k]

  const int tid  = threadIdx.x;
  const int lane = tid & 31;
  const int wave = tid >> 5;
  const int wm   = wave >> 1;      // 0..3 : 32-row slab
  const int wn   = wave & 1;       // 0..1 : 64-col slab
  const int z    = blockIdx.z;

  const __bf16* Ab = A  + (long long)(z / zDiv) * aZ1 + (long long)(z % zDiv) * aZ2;
  const __bf16* Bb = Bt + (long long)(z / zDiv) * bZ1 + (long long)(z % zDiv) * bZ2;
  float*        Cb = C  + (long long)(z / zDiv) * cZ1 + (long long)(z % zDiv) * cZ2;

  const int m0 = blockIdx.y * 128;
  const int n0 = blockIdx.x * 128;

  v8f acc[2][4] = {};

  const int lr   = lane & 15;
  const int kloA = (lane < 16) ? 0 : 8;    // A fragment K base
  const int kloB = (lane < 16) ? 0 : 8;    // B fragment K base

  for (int k0 = 0; k0 < K; k0 += 32) {
    __syncthreads();   // previous chunk's ds_loads already waited before WMMA
    // ---- async-stage A tile [128m x 32k] and B tile [128n x 32k] ----
#pragma unroll
    for (int i = 0; i < 2; ++i) {
      int chunk = tid + 256 * i;            // 512 chunks of 8 bf16 per tile
      int row = chunk >> 2;                 // 0..127
      int kc  = (chunk & 3) * 8;            // 0,8,16,24
      unsigned lofA = (unsigned)(unsigned long long)&lA[row][kc];
      unsigned long long gaA =
          (unsigned long long)(Ab + (long long)(m0 + row) * lda + (k0 + kc));
      asm volatile("global_load_async_to_lds_b128 %0, %1, off"
                   :: "v"(lofA), "v"(gaA) : "memory");
      unsigned lofB = (unsigned)(unsigned long long)&lB[row][kc];
      unsigned long long gaB =
          (unsigned long long)(Bb + (long long)(n0 + row) * ldbt + (k0 + kc));
      asm volatile("global_load_async_to_lds_b128 %0, %1, off"
                   :: "v"(lofB), "v"(gaB) : "memory");
    }
    asm volatile("s_wait_asynccnt 0x0" ::: "memory");
    __syncthreads();

    // ---- build fragments & WMMA ----
    v16bf afrag[2], bfrag[4];
#pragma unroll
    for (int tm = 0; tm < 2; ++tm) {
      const __bf16* base = &lA[wm * 32 + tm * 16 + lr][0];
      v8bf lo = *(const v8bf*)(base + kloA);
      v8bf hi = *(const v8bf*)(base + kloA + 16);
#pragma unroll
      for (int j = 0; j < 8; ++j) { afrag[tm][j] = lo[j]; afrag[tm][j + 8] = hi[j]; }
    }
#pragma unroll
    for (int tn = 0; tn < 4; ++tn) {
      const __bf16* base = &lB[wn * 64 + tn * 16 + lr][0];
      v8bf lo = *(const v8bf*)(base + kloB);
      v8bf hi = *(const v8bf*)(base + kloB + 16);
#pragma unroll
      for (int j = 0; j < 8; ++j) { bfrag[tn][j] = lo[j]; bfrag[tn][j + 8] = hi[j]; }
    }
#pragma unroll
    for (int tm = 0; tm < 2; ++tm)
#pragma unroll
      for (int tn = 0; tn < 4; ++tn)
        acc[tm][tn] = __builtin_amdgcn_wmma_f32_16x16x32_bf16(
            false, afrag[tm], false, bfrag[tn], (short)0, acc[tm][tn],
            false, false);
  }

  // ---- epilogue: C/D layout = VGPR r -> M=(lane<16 ? r : r+8), N=lane%16 ----
#pragma unroll
  for (int tm = 0; tm < 2; ++tm) {
#pragma unroll
    for (int tn = 0; tn < 4; ++tn) {
      int col = n0 + wn * 64 + tn * 16 + lr;
      float bv = bias ? bias[col] : 0.f;
#pragma unroll
      for (int r = 0; r < 8; ++r) {
        int row = m0 + wm * 32 + tm * 16 + ((lane < 16) ? r : r + 8);
        Cb[(long long)row * ldc + col] = acc[tm][tn][r] + bv;
      }
    }
  }
}

// ---------------------------------------------------------------------------
// Host-side orchestration
// ---------------------------------------------------------------------------
static const int Bm = 4, Tm = 4096, Dm = 1024, Lm = 768, Hm = 8, Nm = 512;

extern "C" void kernel_launch(void* const* d_in, const int* in_sizes, int n_in,
                              void* d_out, int out_size, void* d_ws, size_t ws_size,
                              hipStream_t stream) {
  (void)in_sizes; (void)n_in; (void)out_size; (void)ws_size;
  const float* x      = (const float*)d_in[0];   // [B,T,D]
  const float* c      = (const float*)d_in[1];   // [B,N,L]
  const float* ngamma = (const float*)d_in[2];
  const float* nbeta  = (const float*)d_in[3];
  const float* tgamma = (const float*)d_in[4];
  const float* tbeta  = (const float*)d_in[5];
  const float* Wq     = (const float*)d_in[6];
  const float* bq     = (const float*)d_in[7];
  const float* Wk     = (const float*)d_in[8];
  const float* bk     = (const float*)d_in[9];
  const float* Wv     = (const float*)d_in[10];
  const float* bv     = (const float*)d_in[11];
  float* out = (float*)d_out;

  const long long MT = (long long)Bm * Tm;   // 16384
  const long long MN = (long long)Bm * Nm;   // 2048

  size_t off = 0;
  auto alloc = [&](size_t bytes) {
    void* p = (char*)d_ws + off;
    off += (bytes + 255) & ~(size_t)255;
    return p;
  };
  __bf16* WqT   = (__bf16*)alloc((size_t)Dm * Dm * 2);  // [N=1024][K=1024]
  __bf16* WkT   = (__bf16*)alloc((size_t)Dm * Lm * 2);  // [N=1024][K=768]
  __bf16* WvT   = (__bf16*)alloc((size_t)Dm * Lm * 2);
  __bf16* xn    = (__bf16*)alloc((size_t)MT * Dm * 2);
  __bf16* cn    = (__bf16*)alloc((size_t)MN * Lm * 2);
  float*  qf    = (float*)alloc((size_t)MT * Dm * 4);
  float*  kf    = (float*)alloc((size_t)MN * Dm * 4);
  float*  vf    = (float*)alloc((size_t)MN * Dm * 4);
  __bf16* q_s   = (__bf16*)alloc((size_t)MT * Dm * 2);
  __bf16* kt    = (__bf16*)alloc((size_t)Bm * Hm * 128 * Nm * 2);  // [32][128d][512n]
  __bf16* vt    = (__bf16*)alloc((size_t)Bm * Hm * 128 * Nm * 2);  // [32][128l][512n]
  float*  attnf = (float*)alloc((size_t)Bm * Hm * 128 * 128 * 4);  // [32][128d][128l]
  __bf16* attnT = (__bf16*)alloc((size_t)Bm * Hm * 128 * 128 * 2); // [32][128l][128d]

  // 1) weights -> bf16, transposed to [N][K]
  transpose_f32_to_bf16_kernel<<<(Dm * Dm) / 256, 256, 0, stream>>>(Wq, WqT, Dm, Dm);
  transpose_f32_to_bf16_kernel<<<(Lm * Dm) / 256, 256, 0, stream>>>(Wk, WkT, Lm, Dm);
  transpose_f32_to_bf16_kernel<<<(Lm * Dm) / 256, 256, 0, stream>>>(Wv, WvT, Lm, Dm);

  // 2) layernorms -> bf16 activations
  layernorm_bf16_kernel<<<(int)MT, 256, 0, stream>>>(x, ngamma, nbeta, xn, Dm);
  layernorm_bf16_kernel<<<(int)MN, 256, 0, stream>>>(c, tgamma, tbeta, cn, Lm);

  // 3) projections (fp32 out)
  gemm_bf16_nt_kernel<<<dim3(Dm / 128, (int)(MT / 128), 1), 256, 0, stream>>>(
      xn, WqT, qf, bq, Dm, Dm, Dm, Dm, 1, 0, 0, 0, 0, 0, 0);
  gemm_bf16_nt_kernel<<<dim3(Dm / 128, (int)(MN / 128), 1), 256, 0, stream>>>(
      cn, WkT, kf, bk, Lm, Lm, Lm, Dm, 1, 0, 0, 0, 0, 0, 0);
  gemm_bf16_nt_kernel<<<dim3(Dm / 128, (int)(MN / 128), 1), 256, 0, stream>>>(
      cn, WvT, vf, bv, Lm, Lm, Lm, Dm, 1, 0, 0, 0, 0, 0, 0);

  // 4) softmaxes / transposed bf16 operands
  softmax_hd_kernel<<<(int)(MT * Hm / 8), 256, 0, stream>>>(qf, q_s);  // over hd=128
  softmax_col_kernel<<<dim3(Dm / 256, Bm), 256, 0, stream>>>(kf, kt);  // over N + T
  v_transpose_kernel<<<(int)((MN * Dm) / 256), 256, 0, stream>>>(vf, vt);

  // 5) attn[z] = kt[z] * vt[z]^T : z = b*8+h, M=128(d), N=128(l), K=512(n)
  gemm_bf16_nt_kernel<<<dim3(1, 1, Bm * Hm), 256, 0, stream>>>(
      kt, vt, attnf, nullptr, Nm, Nm, Nm, 128, 1,
      (long long)128 * Nm, 0, (long long)128 * Nm, 0,
      (long long)128 * 128, 0);

  // 6) attn -> bf16, transposed to [l][d]
  attn_transpose_kernel<<<(Bm * Hm * 128 * 128) / 256, 256, 0, stream>>>(attnf, attnT);

  // 7) y[b,:,h] = q_s[b,:,h] @ attn[b,h] : M=4096, N=128, K=128
  gemm_bf16_nt_kernel<<<dim3(1, Tm / 128, Bm * Hm), 256, 0, stream>>>(
      q_s, attnT, out, nullptr, 128, Dm, 128, Dm, Hm,
      (long long)Tm * Dm, 128, (long long)Hm * 128 * 128, 128 * 128,
      (long long)Tm * Dm, 128);
}
